// Semilinear_154618823091
// MI455X (gfx1250) — matrix operand order repaired and checked by reading, not compiled, not run on hardware
//
#include <hip/hip_runtime.h>
#include <hip/hip_bf16.h>
#include <stdint.h>

// ---- problem constants (from reference) ----
#define NB     4
#define NL     2048
#define NDIM   1024
#define NH     16
#define NDI    4       // D_INNER
#define NCHUNK 16
#define NKS    4
#define NHD    64
#define NSTEPS (NL / NCHUNK)   // 128
#define NM     (NB * NL)       // 8192 rows
#define NLR    (2 * NH)        // 32

typedef __bf16          v16bf __attribute__((ext_vector_type(16)));
typedef float           v8f   __attribute__((ext_vector_type(8)));
typedef unsigned short  us8   __attribute__((ext_vector_type(8)));

#if defined(__gfx1250__) && __has_builtin(__builtin_amdgcn_wmma_f32_16x16x4_f32)
#define HAVE_WMMA_F32X4 1
typedef float v2f __attribute__((ext_vector_type(2)));
#endif

#if defined(__gfx1250__) && __has_builtin(__builtin_amdgcn_global_load_async_to_lds_b128)
#define HAVE_ASYNC_LDS 1
// builtin signature is (v4i AS1*, v4i AS3*, imm offset, imm cpol) with GCC-style vectors
typedef int v4i_gcc __attribute__((vector_size(16)));
typedef __attribute__((address_space(1))) v4i_gcc* as1_v4i;
typedef __attribute__((address_space(3))) v4i_gcc* as3_v4i;
#endif

__device__ __forceinline__ unsigned short f32_to_bf16(float f) {
  unsigned u = __float_as_uint(f);
  u += 0x7FFFu + ((u >> 16) & 1u);          // RNE
  return (unsigned short)(u >> 16);
}

// 16-byte global -> LDS stage; async DMA (ASYNCcnt) on CDNA5, plain copy fallback
__device__ __forceinline__ void stage16(const unsigned short* g, unsigned short* l) {
#ifdef HAVE_ASYNC_LDS
  __builtin_amdgcn_global_load_async_to_lds_b128((as1_v4i)g, (as3_v4i)l, 0, 0);
#else
  *(us8*)l = *(const us8*)g;
#endif
}

__device__ __forceinline__ void wait_async0() {
#ifdef HAVE_ASYNC_LDS
#if __has_builtin(__builtin_amdgcn_s_wait_asynccnt)
  __builtin_amdgcn_s_wait_asynccnt(0);
#else
  asm volatile("s_wait_asynccnt 0x0" ::: "memory");
#endif
#endif
}

// ---------------- fp32 -> bf16 conversion (grid-stride) ----------------
__global__ __launch_bounds__(256) void cvt_bf16_kernel(const float* __restrict__ src,
                                                       unsigned short* __restrict__ dst,
                                                       int n) {
  int i = blockIdx.x * blockDim.x + threadIdx.x;
  int stride = gridDim.x * blockDim.x;
  for (; i < n; i += stride) dst[i] = f32_to_bf16(src[i]);
}

// ------------- fp32 [K,N] -> bf16 transposed [N,K] (LDS-tiled, coalesced) -------------
__global__ __launch_bounds__(256) void cvt_bf16_T_kernel(const float* __restrict__ src,
                                                         unsigned short* __restrict__ dst,
                                                         int Kd, int Nd) {
  __shared__ float tile[32][33];
  const int bn = blockIdx.x * 32;       // n-base
  const int bk = blockIdx.y * 32;       // k-base
  const int tx = threadIdx.x & 31, ty = threadIdx.x >> 5;   // 32 x 8
#pragma unroll
  for (int i = ty; i < 32; i += 8)
    tile[i][tx] = src[(size_t)(bk + i) * Nd + bn + tx];     // coalesced read rows k
  __syncthreads();
#pragma unroll
  for (int i = ty; i < 32; i += 8)
    dst[(size_t)(bn + i) * Kd + bk + tx] = f32_to_bf16(tile[tx][i]);  // coalesced write rows n
}

// ---------------- bf16 WMMA GEMM: C[M,N](f32) = A[M,K](bf16) @ Wt[N,K](bf16)^T ----------------
// Block: 256 threads = 8 waves. Block tile 128(M) x 64(N); each wave owns a 16x64 strip
// (4 accumulators, 4 WMMAs per 32-K step reusing one A fragment). B comes pre-transposed
// ([N,K]) so the LDS tile is a straight copy -> staged with GLOBAL_LOAD_ASYNC_TO_LDS_B128
// into a double buffer, overlapped with WMMA; one s_wait_asynccnt + one barrier per K-step.
union AFrag { struct { us8 lo, hi; } h; v16bf v; };

__global__ __launch_bounds__(256) void gemm_bf16_wmma_kernel(
    const unsigned short* __restrict__ A,    // [M,K]
    const unsigned short* __restrict__ Wt,   // [N,K]  (transposed weights)
    float* __restrict__ C,                   // [M,N]
    int M, int N, int K) {
  __shared__ __align__(16) unsigned short Bt[2][64 * 40];  // [buf][n][k], k-stride 32->40

  const int tid   = threadIdx.x;
  const int lane  = tid & 31;
  const int wave  = tid >> 5;           // 0..7 -> 16-row strip

  const int m0    = blockIdx.x * 128 + wave * 16;
  const int n0blk = blockIdx.y * 64;

  const int lhalf = lane >> 4;          // 0/1: which K half-group this lane holds
  const int l16   = lane & 15;          // row (A) / col (B,C) within tile

  const unsigned short* arow = A + (size_t)(m0 + l16) * K;

  // B staging: thread -> 16 bytes: n = tid>>2 in [0,64), kq = (tid&3)*8 in {0,8,16,24}
  const int sn_ = tid >> 2;
  const int kq_ = (tid & 3) * 8;
  const unsigned short* brow = Wt + (size_t)(n0blk + sn_) * K + kq_;

  v8f acc0 = {}, acc1 = {}, acc2 = {}, acc3 = {};

  stage16(brow, &Bt[0][sn_ * 40 + kq_]);     // prologue: K-step 0 into buffer 0
  wait_async0();
  __syncthreads();

  for (int k0 = 0; k0 < K; k0 += 32) {
    const int cur = (k0 >> 5) & 1;
    if (k0 + 32 < K) {  // async-stage next K-step into the other buffer while computing
      stage16(brow + k0 + 32, &Bt[cur ^ 1][sn_ * 40 + kq_]);
      __builtin_prefetch(arow + k0 + 64, 0, 1);
    }

    AFrag a, b;
    a.h.lo = *(const us8*)(arow + k0 + lhalf * 16);
    a.h.hi = *(const us8*)(arow + k0 + lhalf * 16 + 8);

    const unsigned short* bbase = &Bt[cur][l16 * 40 + lhalf * 16];
    b.h.lo = *(const us8*)(bbase);            b.h.hi = *(const us8*)(bbase + 8);
    acc0 = __builtin_amdgcn_wmma_f32_16x16x32_bf16(false, a.v, false, b.v, (short)0, acc0, false, false);
    b.h.lo = *(const us8*)(bbase + 16 * 40);  b.h.hi = *(const us8*)(bbase + 16 * 40 + 8);
    acc1 = __builtin_amdgcn_wmma_f32_16x16x32_bf16(false, a.v, false, b.v, (short)0, acc1, false, false);
    b.h.lo = *(const us8*)(bbase + 32 * 40);  b.h.hi = *(const us8*)(bbase + 32 * 40 + 8);
    acc2 = __builtin_amdgcn_wmma_f32_16x16x32_bf16(false, a.v, false, b.v, (short)0, acc2, false, false);
    b.h.lo = *(const us8*)(bbase + 48 * 40);  b.h.hi = *(const us8*)(bbase + 48 * 40 + 8);
    acc3 = __builtin_amdgcn_wmma_f32_16x16x32_bf16(false, a.v, false, b.v, (short)0, acc3, false, false);

    wait_async0();      // our async writes to Bt[cur^1] are done before the fence
    __syncthreads();    // all waves' staging visible; safe to flip buffers
  }

  // C layout: VGPR r -> (M = r + lhalf*8, N = l16) per 16x16 tile
#pragma unroll
  for (int r = 0; r < 8; ++r) {
    const size_t row = (size_t)(m0 + lhalf * 8 + r) * N + n0blk + l16;
    C[row]      = acc0[r];
    C[row + 16] = acc1[r];
    C[row + 32] = acc2[r];
    C[row + 48] = acc3[r];
  }
}

// ---------------- lr projection: hs[M,1024] @ Wlr[1024,32] (raw, f32) ----------------
__global__ __launch_bounds__(256) void lrproj_kernel(const float* __restrict__ hs,
                                                     const float* __restrict__ Wlr,
                                                     float* __restrict__ xlr) {
  int idx = blockIdx.x * blockDim.x + threadIdx.x;   // NM*NLR threads
  int row = idx >> 5, j = idx & 31;
  const float* a = hs + (size_t)row * NDIM;
  float s = 0.f;
#pragma unroll 8
  for (int k = 0; k < NDIM; ++k) s = fmaf(a[k], Wlr[(size_t)k * NLR + j], s);
  xlr[(size_t)row * NLR + j] = s;
}

// ---------------- sequential TTT scan: one workgroup per (b,h), 64 threads ----------------
__global__ __launch_bounds__(64) void scan_kernel(
    const float* __restrict__ xq, const float* __restrict__ xk, const float* __restrict__ xv,
    const float* __restrict__ xlr,
    const float* __restrict__ conv_q, const float* __restrict__ conv_k, const float* __restrict__ conv_v,
    const float* __restrict__ Win_init, const float* __restrict__ Wout_init,
    unsigned short* __restrict__ obf) {
  const int bh = blockIdx.x;
  const int b  = bh / NH, h = bh % NH;
  const int t  = threadIdx.x;          // 0..63
  const int c  = h * NHD + t;          // channel for d-mapped phases

  __shared__ float Win[NDI][NHD], Wout[NDI][NHD];
  __shared__ float qs[NCHUNK][NHD], ks_s[NCHUNK][NHD], vs[NCHUNK][NHD];
  __shared__ float sq_s[NCHUNK][NDI], sk_s[NCHUNK][NDI];
  __shared__ float qh[NCHUNK][NDI], kh[NCHUNK][NDI];
  __shared__ float qkm[NCHUNK][NCHUNK];
  __shared__ float ain[NDI][NCHUNK], aout[NDI][NCHUNK];
  __shared__ float lr0[NCHUNK], lr1[NCHUNK];

#pragma unroll
  for (int D = 0; D < NDI; ++D) {
    Win[D][t]  = Win_init [((size_t)D * NH + h) * NHD + t];
    Wout[D][t] = Wout_init[((size_t)D * NH + h) * NHD + t];
  }
  float cwq[NKS], cwk[NKS], cwv[NKS];
#pragma unroll
  for (int j = 0; j < NKS; ++j) {
    cwq[j] = conv_q[(size_t)c * NKS + j];
    cwk[j] = conv_k[(size_t)c * NKS + j];
    cwv[j] = conv_v[(size_t)c * NKS + j];
  }
  const float scale = 0.125f;          // 1/sqrt(HD)

  for (int n = 0; n < NSTEPS; ++n) {
    const int l0 = n * NCHUNK;

    // -- load + causal depthwise conv + residual (thread t = d) --
    for (int l = 0; l < NCHUNK; ++l) {
      const int gl = l0 + l;
      const size_t base = ((size_t)b * NL + gl) * NDIM + c;
      float yq = xq[base], yk = xk[base], yv = xv[base];
#pragma unroll
      for (int j = 0; j < NKS; ++j) {
        const int ll = gl - (NKS - 1) + j;
        if (ll >= 0) {
          const size_t bb = ((size_t)b * NL + ll) * NDIM + c;
          yq = fmaf(cwq[j], xq[bb], yq);
          yk = fmaf(cwk[j], xk[bb], yk);
          yv = fmaf(cwv[j], xv[bb], yv);
        }
      }
      qs[l][t] = yq; ks_s[l][t] = yk; vs[l][t] = yv;
    }
    if (t < 32) {  // lr = softplus(raw + BASE_LR)
      const int l = t & 15, ch = t >> 4;
      float raw = xlr[((size_t)b * NL + l0 + l) * NLR + 2 * h + ch] + 1.0e-3f;
      float sp  = raw > 20.f ? raw : log1pf(expf(raw));
      if (ch == 0) lr0[l] = sp; else lr1[l] = sp;
    }
    __syncthreads();

    // -- hidden scores s = x @ W_in^T  (thread -> (l,D)) --
    {
      const int l = t >> 2, D = t & 3;
      float aq = 0.f, ak = 0.f;
      for (int d = 0; d < NHD; ++d) {
        aq = fmaf(qs[l][d],   Win[D][d], aq);
        ak = fmaf(ks_s[l][d], Win[D][d], ak);
      }
      sq_s[l][D] = aq; sk_s[l][D] = ak;
    }
    __syncthreads();

    // -- softmax over D=4; k_h gets lr channel-1 gate --
    {
      const int l = t >> 2, D = t & 3;
      float m1 = fmaxf(fmaxf(sq_s[l][0], sq_s[l][1]), fmaxf(sq_s[l][2], sq_s[l][3]));
      float d1 = expf(sq_s[l][0]-m1)+expf(sq_s[l][1]-m1)+expf(sq_s[l][2]-m1)+expf(sq_s[l][3]-m1);
      qh[l][D] = expf(sq_s[l][D] - m1) / d1;
      float m2 = fmaxf(fmaxf(sk_s[l][0], sk_s[l][1]), fmaxf(sk_s[l][2], sk_s[l][3]));
      float d2 = expf(sk_s[l][0]-m2)+expf(sk_s[l][1]-m2)+expf(sk_s[l][2]-m2)+expf(sk_s[l][3]-m2);
      kh[l][D] = expf(sk_s[l][D] - m2) / d2 * lr1[l];
    }
    __syncthreads();

#ifdef HAVE_WMMA_F32X4
    // -- qk = tril(q_h @ k_h^T): single V_WMMA_F32_16X16X4_F32 --
    {
      const int lane = t & 31;
      const int lh = lane >> 4, ln = lane & 15;
      v2f a, bb; v8f acc = {};
      a[0]  = qh[ln][2 * lh];  a[1]  = qh[ln][2 * lh + 1];
      bb[0] = kh[ln][2 * lh];  bb[1] = kh[ln][2 * lh + 1];
      acc = __builtin_amdgcn_wmma_f32_16x16x4_f32(false, a, false, bb, (short)0, acc, false, false);
      if (t < 32) {  // wave0 stores the masked tile (both waves ran the MMA, full EXEC)
#pragma unroll
        for (int r = 0; r < 8; ++r) {
          const int mrow = r + lh * 8;
          qkm[mrow][ln] = (ln <= mrow) ? acc[r] : 0.f;
        }
      }
    }
#else
    {
      const int qi = t >> 2;
#pragma unroll
      for (int jj = 0; jj < 4; ++jj) {
        const int ki = (t & 3) * 4 + jj;
        float s = 0.f;
        if (ki <= qi) {
#pragma unroll
          for (int D = 0; D < NDI; ++D) s = fmaf(qh[qi][D], kh[ki][D], s);
        }
        qkm[qi][ki] = s;
      }
    }
#endif
    __syncthreads();

#ifdef HAVE_WMMA_F32X4
    // -- o = q_h @ W_out(old) + qk @ v : 5 chained K=4 WMMAs per 16-col tile --
    {
      const int lane = t & 31, w = t >> 5;   // wave0 -> cols 0..31, wave1 -> 32..63
      const int lh = lane >> 4, ln = lane & 15;
#pragma unroll
      for (int tt = 0; tt < 2; ++tt) {
        const int n0 = (w * 2 + tt) * 16;
        v8f acc = {}; v2f a, bb;
        a[0]  = qh[ln][2 * lh];        a[1]  = qh[ln][2 * lh + 1];
        bb[0] = Wout[2 * lh][n0 + ln]; bb[1] = Wout[2 * lh + 1][n0 + ln];
        acc = __builtin_amdgcn_wmma_f32_16x16x4_f32(false, a, false, bb, (short)0, acc, false, false);
#pragma unroll
        for (int ks0 = 0; ks0 < NCHUNK; ks0 += 4) {
          a[0]  = qkm[ln][ks0 + 2 * lh];      a[1]  = qkm[ln][ks0 + 2 * lh + 1];
          bb[0] = vs[ks0 + 2 * lh][n0 + ln];  bb[1] = vs[ks0 + 2 * lh + 1][n0 + ln];
          acc = __builtin_amdgcn_wmma_f32_16x16x4_f32(false, a, false, bb, (short)0, acc, false, false);
        }
#pragma unroll
        for (int r = 0; r < 8; ++r) {
          const int mrow = r + lh * 8;
          obf[((size_t)b * NL + l0 + mrow) * NDIM + h * NHD + n0 + ln] = f32_to_bf16(acc[r]);
        }
      }
    }
#else
    for (int l = 0; l < NCHUNK; ++l) {
      float s = 0.f;
#pragma unroll
      for (int D = 0; D < NDI; ++D) s = fmaf(qh[l][D], Wout[D][t], s);
      for (int k2 = 0; k2 <= l; ++k2) s = fmaf(qkm[l][k2], vs[k2][t], s);
      obf[((size_t)b * NL + l0 + l) * NDIM + c] = f32_to_bf16(s);
    }
#endif
    __syncthreads();  // o read Wout across all columns; fence before per-column update

    // -- W_out += k_h^T @ v (thread t = d) --
#pragma unroll
    for (int D = 0; D < NDI; ++D) {
      float s = 0.f;
      for (int l = 0; l < NCHUNK; ++l) s = fmaf(kh[l][D], vs[l][t], s);
      Wout[D][t] += s;
    }
    __syncthreads();

    // -- two inner TTT gradient steps --
    for (int it = 0; it < 2; ++it) {
      {  // logits (thread -> (D,k)): in-head = W_in.k ; out-head = W_out.v
        const int D = t >> 4, k2 = t & 15;
        float zin = 0.f, zout = 0.f;
        for (int d = 0; d < NHD; ++d) {
          zin  = fmaf(Win[D][d],  ks_s[k2][d], zin);
          zout = fmaf(Wout[D][d], vs[k2][d],   zout);
        }
        ain[D][k2]  = zin * scale;
        aout[D][k2] = zout * scale;
      }
      __syncthreads();
      float vin_n, vout_n;
      {  // softmax over k=16 (into regs; write after barrier)
        const int D = t >> 4, k2 = t & 15;
        float m1 = -3.0e38f, m2 = -3.0e38f;
#pragma unroll
        for (int i = 0; i < NCHUNK; ++i) { m1 = fmaxf(m1, ain[D][i]); m2 = fmaxf(m2, aout[D][i]); }
        float s1 = 0.f, s2 = 0.f;
#pragma unroll
        for (int i = 0; i < NCHUNK; ++i) { s1 += expf(ain[D][i] - m1); s2 += expf(aout[D][i] - m2); }
        vin_n  = expf(ain[D][k2]  - m1) / s1;
        vout_n = expf(aout[D][k2] - m2) / s2;
      }
      __syncthreads();
      { const int D = t >> 4, k2 = t & 15; ain[D][k2] = vin_n; aout[D][k2] = vout_n; }
      __syncthreads();
      // updates: W_in += lr0[pos0]*(attn_out @ k) ; W_out += lr1[pos0]*(attn_in @ v)
      const float g0 = lr0[0], g1 = lr1[0];
#pragma unroll
      for (int D = 0; D < NDI; ++D) {
        float si = 0.f, so = 0.f;
        for (int l = 0; l < NCHUNK; ++l) {
          si = fmaf(aout[D][l], ks_s[l][t], si);
          so = fmaf(ain[D][l],  vs[l][t],  so);
        }
        Win[D][t]  += g0 * si;
        Wout[D][t] += g1 * so;
      }
      __syncthreads();
    }
  }
}

// ---------------- host launch ----------------
extern "C" void kernel_launch(void* const* d_in, const int* in_sizes, int n_in,
                              void* d_out, int out_size, void* d_ws, size_t ws_size,
                              hipStream_t stream) {
  (void)in_sizes; (void)n_in; (void)out_size; (void)ws_size;
  const float* hs    = (const float*)d_in[0];
  const float* Wq    = (const float*)d_in[1];
  const float* Wk    = (const float*)d_in[2];
  const float* Wv    = (const float*)d_in[3];
  const float* Wlr   = (const float*)d_in[4];
  const float* Wo    = (const float*)d_in[5];
  const float* cq    = (const float*)d_in[6];
  const float* ck    = (const float*)d_in[7];
  const float* cv    = (const float*)d_in[8];
  const float* Wini  = (const float*)d_in[9];
  const float* Wouti = (const float*)d_in[10];
  float* out = (float*)d_out;

  // workspace carve (~122 MB): hs_bf16 reused for o_bf16 after projection GEMMs
  char* ws = (char*)d_ws;
  size_t off = 0;
  auto take = [&](size_t bytes) { char* p = ws + off; off += (bytes + 255) & ~(size_t)255; return p; };
  unsigned short* hsb = (unsigned short*)take((size_t)NM * NDIM * 2);
  unsigned short* wqb = (unsigned short*)take((size_t)NDIM * NDIM * 2);   // transposed [N,K]
  unsigned short* wkb = (unsigned short*)take((size_t)NDIM * NDIM * 2);
  unsigned short* wvb = (unsigned short*)take((size_t)NDIM * NDIM * 2);
  unsigned short* wob = (unsigned short*)take((size_t)NDIM * NDIM * 2);
  float* xq  = (float*)take((size_t)NM * NDIM * 4);
  float* xk  = (float*)take((size_t)NM * NDIM * 4);
  float* xv  = (float*)take((size_t)NM * NDIM * 4);
  float* xlr = (float*)take((size_t)NM * NLR * 4);
  unsigned short* obf = hsb;  // stream-ordered reuse: hs_bf16 dead after the 3 GEMMs

  const int nhs = NM * NDIM;
  cvt_bf16_kernel<<<1024, 256, 0, stream>>>(hs, hsb, nhs);
  dim3 gt(NDIM / 32, NDIM / 32);
  cvt_bf16_T_kernel<<<gt, 256, 0, stream>>>(Wq, wqb, NDIM, NDIM);
  cvt_bf16_T_kernel<<<gt, 256, 0, stream>>>(Wk, wkb, NDIM, NDIM);
  cvt_bf16_T_kernel<<<gt, 256, 0, stream>>>(Wv, wvb, NDIM, NDIM);
  cvt_bf16_T_kernel<<<gt, 256, 0, stream>>>(Wo, wob, NDIM, NDIM);

  lrproj_kernel<<<(NM * NLR) / 256, 256, 0, stream>>>(hs, Wlr, xlr);

  dim3 gg(NM / 128, NDIM / 64);
  gemm_bf16_wmma_kernel<<<gg, 256, 0, stream>>>(hsb, wqb, xq, NM, NDIM, NDIM);
  gemm_bf16_wmma_kernel<<<gg, 256, 0, stream>>>(hsb, wkb, xk, NM, NDIM, NDIM);
  gemm_bf16_wmma_kernel<<<gg, 256, 0, stream>>>(hsb, wvb, xv, NM, NDIM, NDIM);

  scan_kernel<<<NB * NH, 64, 0, stream>>>(xq, xk, xv, xlr, cq, ck, cv, Wini, Wouti, obf);

  gemm_bf16_wmma_kernel<<<gg, 256, 0, stream>>>(obf, wob, out, NM, NDIM, NDIM);
}
